// TrafficLightDetector_90520730731203
// MI455X (gfx1250) — matrix operand order repaired and checked by reading, not compiled, not run on hardware
//
#include <hip/hip_runtime.h>
#include <cstdint>

// ---------------------------------------------------------------------------
// MI455X (gfx1250) implementation.
// Compute-bound on the two 3x3 256->256 convs (~0.82 TFLOP total vs ~17us of
// HBM traffic at 23.3 TB/s), so everything is built around
// V_WMMA_F32_16X16X32_F16 (f16 inputs, f32 accumulate) with implicit-GEMM
// tiling, NHWC f16 activations (K=ic contiguous -> b128 fragment loads), and
// a double-buffered async global->LDS pipeline (ASYNCcnt) so the next K-chunk
// streams into LDS while the current chunk's 36 WMMAs execute.
// ---------------------------------------------------------------------------

typedef _Float16 v16h __attribute__((ext_vector_type(16)));
typedef _Float16 v8h  __attribute__((ext_vector_type(8)));
typedef float    v8f  __attribute__((ext_vector_type(8)));
typedef int      v4i  __attribute__((ext_vector_type(4)));

#define DEV __device__ __forceinline__

DEV v8f wmma32(v16h a, v16h b, v8f c) {
  // D = A(16x32 f16) * B(32x16 f16) + C(16x16 f32)
  return __builtin_amdgcn_wmma_f32_16x16x32_f16(false, a, false, b,
                                                (short)0, c, false, false);
}

// A-matrix fragment (16x32 f16). Lane L holds row M=L%16;
// K = (L/16)*16 + [0..15] in v16h elems 0..15 (contiguous, matches ldB).
DEV v16h ldA(const _Float16* p, int stride, int lane) {
  const _Float16* q = p + (lane & 15) * stride + ((lane >> 4) << 4);
  v8h lo = *(const v8h*)(q);
  v8h hi = *(const v8h*)(q + 8);
  return __builtin_shufflevector(lo, hi, 0, 1, 2, 3, 4, 5, 6, 7,
                                 8, 9, 10, 11, 12, 13, 14, 15);
}

// B-matrix fragment (32x16 f16). Lane L holds column N=L%16;
// K = (L/16)*16 + [0..15] -> 32 contiguous bytes in a [n][k] layout.
DEV v16h ldB(const _Float16* p, int stride, int lane) {
  const _Float16* q = p + (lane & 15) * stride + ((lane >> 4) << 4);
  v8h lo = *(const v8h*)(q);
  v8h hi = *(const v8h*)(q + 8);
  return __builtin_shufflevector(lo, hi, 0, 1, 2, 3, 4, 5, 6, 7,
                                 8, 9, 10, 11, 12, 13, 14, 15);
}

// 16B global -> LDS copy; async path (ASYNCcnt) when the builtin exists.
DEV void cp16_g2l(const _Float16* g, _Float16* l) {
#if __has_builtin(__builtin_amdgcn_global_load_async_to_lds_b128)
  __builtin_amdgcn_global_load_async_to_lds_b128(
      (__attribute__((address_space(1))) v4i*)g,
      (__attribute__((address_space(3))) v4i*)l, 0, 0);
#else
  *(float4*)(void*)l = *(const float4*)(const void*)g;
#endif
}

DEV void wait_async3() {   // <=3 outstanding: previous stage fully landed
#if __has_builtin(__builtin_amdgcn_s_wait_asynccnt)
  __builtin_amdgcn_s_wait_asynccnt(3);
#elif defined(__gfx1250__)
  asm volatile("s_wait_asynccnt 3" ::: "memory");
#endif
}
DEV void wait_async0() {
#if __has_builtin(__builtin_amdgcn_s_wait_asynccnt)
  __builtin_amdgcn_s_wait_asynccnt(0);
#elif defined(__gfx1250__)
  asm volatile("s_wait_asynccnt 0" ::: "memory");
#endif
}

// ---------------------------------------------------------------------------
// Weight prep: cast to f16, fold BN scale into 3x3 weights, pad pred M to 48.
// ---------------------------------------------------------------------------

__global__ void cast_w_k(const float* __restrict__ src,
                         _Float16* __restrict__ dst, int n) {
  int i = blockIdx.x * blockDim.x + threadIdx.x;
  if (i < n) dst[i] = (_Float16)src[i];
}

__global__ void zero_h_k(_Float16* __restrict__ dst, int n) {
  int i = blockIdx.x * blockDim.x + threadIdx.x;
  if (i < n) dst[i] = (_Float16)0.f;
}

// w (L,256,256,3,3) OIHW * (g/sqrt(v+eps))[oc]  ->  [L][ky][kx][oc][ic] f16
__global__ void prep3x3_k(const float* __restrict__ w,
                          const float* __restrict__ g,
                          const float* __restrict__ v,
                          _Float16* __restrict__ dst, int n) {
  int i = blockIdx.x * blockDim.x + threadIdx.x;
  if (i >= n) return;
  int ic = i & 255;
  int t2 = i >> 8;
  int oc = t2 & 255;
  int t3 = t2 >> 8;
  int kk = t3 % 9;
  int l  = t3 / 9;
  float s = g[l * 256 + oc] * rsqrtf(v[l * 256 + oc] + 1e-5f);
  dst[i] = (_Float16)(w[((l * 256 + oc) * 256 + ic) * 9 + kk] * s);
}

__global__ void prep_bias_k(const float* __restrict__ b,
                            const float* __restrict__ g,
                            const float* __restrict__ be,
                            const float* __restrict__ m,
                            const float* __restrict__ v,
                            float* __restrict__ dst, int n) {
  int i = blockIdx.x * blockDim.x + threadIdx.x;
  if (i >= n) return;
  float s = g[i] * rsqrtf(v[i] + 1e-5f);
  dst[i] = (b[i] - m[i]) * s + be[i];
}

// wp (L,45,256) -> [L][48][256] f16, rows 45..47 zero
__global__ void prep_pred_k(const float* __restrict__ wp,
                            _Float16* __restrict__ dst, int n) {
  int i = blockIdx.x * blockDim.x + threadIdx.x;
  if (i >= n) return;
  int ic = i & 255;
  int oc = (i >> 8) % 48;
  int l  = i / (48 * 256);
  dst[i] = (oc < 45) ? (_Float16)wp[(l * 45 + oc) * 256 + ic] : (_Float16)0.f;
}

// ---------------------------------------------------------------------------
// adapt: 1x1 conv 256->256, NCHW f32 in -> NHWC f16 out (+bias, no act).
// Block = 8 waves: 4 M-groups x 2 N-groups; tile M=128(grid.y=2), N=64 px.
// ---------------------------------------------------------------------------

__global__ __launch_bounds__(256) void adapt1x1_k(
    const float* __restrict__ feat,     // [256][HW] one image
    const _Float16* __restrict__ wf,    // [256][256] f16
    const float* __restrict__ bias,     // [256]
    _Float16* __restrict__ out,         // [HW][256] f16
    int HW) {
  __shared__ __align__(16) _Float16 sX[64 * 32];  // [px][ic] transposed stage
  const int t = threadIdx.x;
  const int lane = t & 31, wv = t >> 5;
  const int mwv = wv & 3, nwv = wv >> 2;
  const int p0  = blockIdx.x * 64;
  const int oc0 = blockIdx.y * 128 + mwv * 32;
  const int nb  = nwv * 32;

  v8f acc[2][2] = {};

  for (int icc = 0; icc < 8; ++icc) {
#pragma unroll
    for (int j = 0; j < 8; ++j) {       // coalesced f32 loads, f16 LDS scatter
      int ic = j * 4 + (t >> 6);
      int p  = t & 63;
      sX[p * 32 + ic] = (_Float16)feat[(size_t)(icc * 32 + ic) * HW + p0 + p];
    }
    __syncthreads();
    const _Float16* wb = wf + (size_t)oc0 * 256 + icc * 32;
    v16h a0 = ldA(wb, 256, lane);
    v16h a1 = ldA(wb + 16 * 256, 256, lane);
    v16h b0 = ldB(sX + nb * 32, 32, lane);
    v16h b1 = ldB(sX + (nb + 16) * 32, 32, lane);
    acc[0][0] = wmma32(a0, b0, acc[0][0]);
    acc[0][1] = wmma32(a0, b1, acc[0][1]);
    acc[1][0] = wmma32(a1, b0, acc[1][0]);
    acc[1][1] = wmma32(a1, b1, acc[1][1]);
    __syncthreads();
  }

  const int hi = (lane >> 4) << 3;
#pragma unroll
  for (int mt = 0; mt < 2; ++mt)
#pragma unroll
    for (int nt = 0; nt < 2; ++nt) {
      int px  = p0 + nb + nt * 16 + (lane & 15);
      int ocf = oc0 + mt * 16 + hi;
      v8h pk;
#pragma unroll
      for (int e = 0; e < 8; ++e)
        pk[e] = (_Float16)(acc[mt][nt][e] + bias[ocf + e]);
      *(v8h*)(out + (size_t)px * 256 + ocf) = pk;   // oc-contiguous b128 store
    }
}

// ---------------------------------------------------------------------------
// 3x3 conv 256->256 + folded-BN bias + ReLU, NHWC f16 -> NHWC f16.
// Implicit GEMM, K = 9*256. Per WG: one output row y, 64 x-pixels, 128 oc.
// Double-buffered LDS: stage(icc+1) is issued async, then we wait
// ASYNCcnt<=3 (in-order completion => all of stage(icc) has landed) and run
// the 9 taps x 4 WMMAs of chunk icc while chunk icc+1 streams in.
// OOB halo chunks async-copy from a global zero page so every wave issues a
// fixed <=4 async ops per stage (makes the <=3 wait threshold valid).
// ---------------------------------------------------------------------------

DEV void stage_chunk(int q, const _Float16* __restrict__ in,
                     const _Float16* __restrict__ zpad,
                     _Float16* sbuf, int y, int x0, int H, int W, int icc) {
  int r    = q / 264;          // row 0..2
  int rem  = q - r * 264;
  int p    = rem >> 2;         // halo pixel 0..65 (x = x0 + p - 1)
  int part = rem & 3;          // 16B quarter of the 64B ic-chunk
  int yg = y + r - 1;
  int xg = x0 + p - 1;
  bool ok = (unsigned)yg < (unsigned)H && (unsigned)xg < (unsigned)W;
  const _Float16* src =
      ok ? in + ((size_t)(yg * W + xg) * 256 + icc * 32 + part * 8)
         : zpad + part * 8;    // branchless: padding copies zeros
  stage_chunk_store:
  cp16_g2l(src, &sbuf[(r * 68 + p) * 32 + part * 8]);
}

DEV void stage3x3(const _Float16* __restrict__ in,
                  const _Float16* __restrict__ zpad, _Float16* sbuf,
                  int t, int y, int x0, int H, int W, int icc) {
  // 3 rows x 66 px x 4 quarters = 792 16B chunks; 3 full passes + 24 extras.
#pragma unroll
  for (int pass = 0; pass < 3; ++pass)
    stage_chunk(t + pass * 256, in, zpad, sbuf, y, x0, H, W, icc);
  if (t < 24)
    stage_chunk(t + 768, in, zpad, sbuf, y, x0, H, W, icc);
}

DEV void compute3x3(const _Float16* __restrict__ wf, const _Float16* sbuf,
                    int lane, int oc0, int nb, int icc, v8f acc[2][2]) {
#pragma unroll
  for (int ky = 0; ky < 3; ++ky) {
#pragma unroll
    for (int kx = 0; kx < 3; ++kx) {
      const _Float16* wb =
          wf + ((size_t)(ky * 3 + kx) * 256 + oc0) * 256 + icc * 32;
      v16h a0 = ldA(wb, 256, lane);
      v16h a1 = ldA(wb + 16 * 256, 256, lane);
      const _Float16* bb = sbuf + (ky * 68 + nb + kx) * 32;
      v16h b0 = ldB(bb, 32, lane);
      v16h b1 = ldB(bb + 16 * 32, 32, lane);
      acc[0][0] = wmma32(a0, b0, acc[0][0]);
      acc[0][1] = wmma32(a0, b1, acc[0][1]);
      acc[1][0] = wmma32(a1, b0, acc[1][0]);
      acc[1][1] = wmma32(a1, b1, acc[1][1]);
    }
  }
}

__global__ __launch_bounds__(256) void conv3x3_k(
    const _Float16* __restrict__ in,    // [H*W][256] f16
    const _Float16* __restrict__ wf,    // [9][256][256] f16 (BN-folded)
    const float* __restrict__ bias,     // [256] folded
    const _Float16* __restrict__ zpad,  // >=64B of f16 zeros (global)
    _Float16* __restrict__ out,         // [H*W][256] f16
    int H, int W) {
  __shared__ __align__(16) _Float16 sB[2][3 * 68 * 32];  // double buffer
  const int t = threadIdx.x;
  const int lane = t & 31, wv = t >> 5;
  const int mwv = wv & 3, nwv = wv >> 2;
  const int x0  = blockIdx.x * 64;
  const int y   = blockIdx.z;
  const int oc0 = blockIdx.y * 128 + mwv * 32;
  const int nb  = nwv * 32;

  v8f acc[2][2] = {};

  stage3x3(in, zpad, sB[0], t, y, x0, H, W, 0);       // prologue
  for (int icc = 0; icc < 7; ++icc) {
    stage3x3(in, zpad, sB[(icc + 1) & 1], t, y, x0, H, W, icc + 1);
    wait_async3();            // stage(icc) landed; stage(icc+1) in flight
    __syncthreads();
    compute3x3(wf, sB[icc & 1], lane, oc0, nb, icc, acc);
    __syncthreads();
  }
  wait_async0();
  __syncthreads();
  compute3x3(wf, sB[1], lane, oc0, nb, 7, acc);       // epilogue chunk

  const int hi = (lane >> 4) << 3;
#pragma unroll
  for (int mt = 0; mt < 2; ++mt)
#pragma unroll
    for (int nt = 0; nt < 2; ++nt) {
      int px = x0 + nb + nt * 16 + (lane & 15);
      if (px >= W) continue;                       // x-tail (W=160/80/40)
      int ocf = oc0 + mt * 16 + hi;
      v8h pk;
#pragma unroll
      for (int e = 0; e < 8; ++e) {
        float r = acc[mt][nt][e] + bias[ocf + e];
        pk[e] = (_Float16)fmaxf(r, 0.f);           // fused ReLU
      }
      *(v8h*)(out + (size_t)(y * W + px) * 256 + ocf) = pk;
    }
}

// ---------------------------------------------------------------------------
// pred: 1x1 conv 256->45 (M padded to 48) + activations + scatter to d_out.
// NHWC input is K-contiguous, so B fragments load straight from global.
// ---------------------------------------------------------------------------

__global__ __launch_bounds__(256) void pred_k(
    const _Float16* __restrict__ x,     // [HW][256] f16
    const _Float16* __restrict__ wf,    // [48][256] f16
    const float* __restrict__ bias,     // [45]
    float* __restrict__ out, int outbase, int HW, int b) {
  const int t = threadIdx.x, lane = t & 31, wv = t >> 5;
  const int pxb = blockIdx.x * 256 + wv * 32;

  v8f acc[3][2] = {};

#pragma unroll
  for (int icc = 0; icc < 8; ++icc) {
    v16h bfr[2];
#pragma unroll
    for (int nt = 0; nt < 2; ++nt) {
      int px = pxb + nt * 16 + (lane & 15);
      px = px < HW ? px : HW - 1;                  // clamp tail loads
      const _Float16* q = x + (size_t)px * 256 + icc * 32 + ((lane >> 4) << 4);
      v8h lo = *(const v8h*)q;
      v8h hi = *(const v8h*)(q + 8);
      bfr[nt] = __builtin_shufflevector(lo, hi, 0, 1, 2, 3, 4, 5, 6, 7,
                                        8, 9, 10, 11, 12, 13, 14, 15);
    }
#pragma unroll
    for (int mt = 0; mt < 3; ++mt) {
      v16h a = ldA(wf + (size_t)(mt * 16) * 256 + icc * 32, 256, lane);
      acc[mt][0] = wmma32(a, bfr[0], acc[mt][0]);
      acc[mt][1] = wmma32(a, bfr[1], acc[mt][1]);
    }
  }

  const int hi8 = (lane >> 4) << 3;
#pragma unroll
  for (int mt = 0; mt < 3; ++mt)
#pragma unroll
    for (int nt = 0; nt < 2; ++nt) {
      int px = pxb + nt * 16 + (lane & 15);
      if (px >= HW) continue;
#pragma unroll
      for (int e = 0; e < 8; ++e) {
        int oc = mt * 16 + hi8 + e;
        if (oc >= 45) continue;
        float r = acc[mt][nt][e] + bias[oc];
        int a = oc / 15, j = oc - a * 15;
        int off;
        if (j < 4) {                                    // bbox (copy)
          off = b * 12 * HW + (a * 4 + j) * HW;
        } else if (j == 4) {                            // obj (sigmoid)
          r = 1.f / (1.f + expf(-r));
          off = 48 * HW + b * 3 * HW + a * HW;
        } else if (j < 10) {                            // state (copy)
          off = 60 * HW + b * 15 * HW + (a * 5 + (j - 5)) * HW;
        } else if (j < 14) {                            // arrow (copy)
          off = 120 * HW + b * 12 * HW + (a * 4 + (j - 10)) * HW;
        } else {                                        // dist (softplus+1)
          r = (r > 20.f) ? r : log1pf(expf(r));
          r += 1.f;
          off = 168 * HW + b * 3 * HW + a * HW;
        }
        out[outbase + off + px] = r;
      }
    }
}

// ---------------------------------------------------------------------------
// Host launcher.
// ---------------------------------------------------------------------------

extern "C" void kernel_launch(void* const* d_in, const int* in_sizes, int n_in,
                              void* d_out, int out_size, void* d_ws,
                              size_t ws_size, hipStream_t stream) {
  (void)in_sizes; (void)n_in; (void)out_size; (void)ws_size;

  const float* feat[4] = {(const float*)d_in[0], (const float*)d_in[1],
                          (const float*)d_in[2], (const float*)d_in[3]};
  const float* adapt_w = (const float*)d_in[4];
  const float* adapt_b = (const float*)d_in[5];
  const float* w1 = (const float*)d_in[6];
  const float* b1 = (const float*)d_in[7];
  const float* g1 = (const float*)d_in[8];
  const float* be1 = (const float*)d_in[9];
  const float* m1 = (const float*)d_in[10];
  const float* v1 = (const float*)d_in[11];
  const float* w2 = (const float*)d_in[12];
  const float* b2 = (const float*)d_in[13];
  const float* g2 = (const float*)d_in[14];
  const float* be2 = (const float*)d_in[15];
  const float* m2 = (const float*)d_in[16];
  const float* v2 = (const float*)d_in[17];
  const float* wp = (const float*)d_in[18];
  const float* bp = (const float*)d_in[19];

  // Workspace layout (~73 MB total).
  char* ws = (char*)d_ws;
  size_t o = 0;
  auto take = [&](size_t bytes) -> char* {
    char* p = ws + o;
    o += (bytes + 255) & ~(size_t)255;
    return p;
  };
  _Float16* bufA = (_Float16*)take(61440ull * 256 * 2);   // max-level NHWC f16
  _Float16* bufB = (_Float16*)take(61440ull * 256 * 2);
  _Float16* wAf  = (_Float16*)take(4ull * 256 * 256 * 2);
  _Float16* w1f  = (_Float16*)take(4ull * 9 * 256 * 256 * 2);
  _Float16* w2f  = (_Float16*)take(4ull * 9 * 256 * 256 * 2);
  _Float16* wpf  = (_Float16*)take(4ull * 48 * 256 * 2);
  float*    b1f  = (float*)take(1024 * 4);
  float*    b2f  = (float*)take(1024 * 4);
  _Float16* zpad = (_Float16*)take(256);                  // global zero page

  // Weight preparation (BN folding + f16 casts) + zero page.
  zero_h_k<<<1, 128, 0, stream>>>(zpad, 128);
  cast_w_k<<<(262144 + 255) / 256, 256, 0, stream>>>(adapt_w, wAf, 262144);
  prep3x3_k<<<(2359296 + 255) / 256, 256, 0, stream>>>(w1, g1, v1, w1f, 2359296);
  prep3x3_k<<<(2359296 + 255) / 256, 256, 0, stream>>>(w2, g2, v2, w2f, 2359296);
  prep_bias_k<<<4, 256, 0, stream>>>(b1, g1, be1, m1, v1, b1f, 1024);
  prep_bias_k<<<4, 256, 0, stream>>>(b2, g2, be2, m2, v2, b2f, 1024);
  prep_pred_k<<<(49152 + 255) / 256, 256, 0, stream>>>(wp, wpf, 49152);

  static const int Hs[4] = {192, 96, 48, 24};
  static const int Wl[4] = {320, 160, 80, 40};

  int outbase = 0;
  for (int l = 0; l < 4; ++l) {
    const int H = Hs[l], W = Wl[l], HW = H * W;
    for (int b = 0; b < 4; ++b) {
      adapt1x1_k<<<dim3(HW / 64, 2), 256, 0, stream>>>(
          feat[l] + (size_t)b * 256 * HW, wAf + (size_t)l * 65536,
          adapt_b + l * 256, bufA, HW);
      conv3x3_k<<<dim3((W + 63) / 64, 2, H), 256, 0, stream>>>(
          bufA, w1f + (size_t)l * 9 * 65536, b1f + l * 256, zpad, bufB, H, W);
      conv3x3_k<<<dim3((W + 63) / 64, 2, H), 256, 0, stream>>>(
          bufB, w2f + (size_t)l * 9 * 65536, b2f + l * 256, zpad, bufA, H, W);
      pred_k<<<dim3((HW + 255) / 256), 256, 0, stream>>>(
          bufA, wpf + (size_t)l * 48 * 256, bp + l * 45, (float*)d_out,
          outbase, HW, b);
    }
    outbase += 180 * HW;   // 45 channels * B=4 * HW floats per level
  }
}
